// StreamingDeepseekMoE_55009941127245
// MI455X (gfx1250) — matrix operand order repaired and checked by reading, not compiled, not run on hardware
//
#include <hip/hip_runtime.h>
#include <hip/hip_bf16.h>
#include <hip/hip_fp16.h>

#define HID 2048
#define NE 8
#define SCALE_F 2.5f
#define NWAVES 8          // 256 threads / wave32
#define MTILE 32          // tokens per block tile

typedef __attribute__((ext_vector_type(16))) _Float16 v16h;
typedef __attribute__((ext_vector_type(8)))  _Float16 v8h;
typedef __attribute__((ext_vector_type(8)))  float    v8f;

// ---------------- utility kernels ----------------

__global__ void cvt_x_f16(const float* __restrict__ x, _Float16* __restrict__ xh, int n) {
    int i = blockIdx.x * blockDim.x + threadIdx.x;
    if (i < n) xh[i] = (_Float16)x[i];
}

// out[mat][n*K + k] = (f16) in[mat][k*N + n]   (coalesced writes)
__global__ void transpose_cvt(const float* __restrict__ in, _Float16* __restrict__ out,
                              int K, int N) {
    size_t mat = blockIdx.y;
    in  += mat * (size_t)K * N;
    out += mat * (size_t)K * N;
    int i = blockIdx.x * blockDim.x + threadIdx.x;
    if (i < K * N) {
        int n = i / K;
        int k = i - n * K;
        out[i] = (_Float16)in[(size_t)k * N + n];
    }
}

// ---------------- router: softmax + top-2 -> dense combine[T, NE] ----------------

__global__ __launch_bounds__(64) void router_kernel(const float* __restrict__ x,
                                                    const float* __restrict__ wr,
                                                    float* __restrict__ combine) {
    __shared__ float part[64][NE];
    int t   = blockIdx.x;
    int tid = threadIdx.x;
    float acc[NE];
#pragma unroll
    for (int e = 0; e < NE; e++) acc[e] = 0.f;
    const float* xr = x + (size_t)t * HID;
    for (int k = tid; k < HID; k += 64) {
        float xv = xr[k];
        const float* w = wr + (size_t)k * NE;
#pragma unroll
        for (int e = 0; e < NE; e++) acc[e] += xv * w[e];
    }
#pragma unroll
    for (int e = 0; e < NE; e++) part[tid][e] = acc[e];
    __syncthreads();
    if (tid == 0) {
        float lg[NE];
        for (int e = 0; e < NE; e++) {
            float s = 0.f;
            for (int j = 0; j < 64; j++) s += part[j][e];
            lg[e] = s;
        }
        float mx = lg[0];
        for (int e = 1; e < NE; e++) mx = fmaxf(mx, lg[e]);
        float pe[NE], den = 0.f;
        for (int e = 0; e < NE; e++) { pe[e] = __expf(lg[e] - mx); den += pe[e]; }
        for (int e = 0; e < NE; e++) pe[e] /= den;
        int i1 = 0;
        for (int e = 1; e < NE; e++) if (pe[e] > pe[i1]) i1 = e;
        int i2 = (i1 == 0) ? 1 : 0;
        for (int e = 0; e < NE; e++) { if (e == i1) continue; if (pe[e] > pe[i2]) i2 = e; }
        float s  = pe[i1] + pe[i2];
        float w1 = pe[i1] / s, w2 = pe[i2] / s;
        float* cr = combine + (size_t)t * NE;
        for (int e = 0; e < NE; e++) cr[e] = (e == i1) ? w1 : ((e == i2) ? w2 : 0.f);
    }
}

__global__ void build_lists(const float* __restrict__ combine, int* __restrict__ counts,
                            int* __restrict__ lists, int T) {
    int t = blockIdx.x * blockDim.x + threadIdx.x;
    if (t >= T) return;
#pragma unroll
    for (int e = 0; e < NE; e++) {
        if (combine[(size_t)t * NE + e] > 0.f) {
            int pos = atomicAdd(&counts[e], 1);
            lists[e * T + pos] = t;
        }
    }
}

// ---------------- fused SwiGLU expert (WMMA), 32-token tile ----------------
// A (16x32 f16): row = lane&15; lanes<16 hold K {0-15}, lanes>=16 {16-31}
__device__ inline v16h load_a16(const _Float16* row) {
    v8h lo = *(const v8h*)(row);
    v8h hi = *(const v8h*)(row + 8);
    return __builtin_shufflevector(lo, hi, 0,1,2,3,4,5,6,7,8,9,10,11,12,13,14,15);
}

#define WMMA16(A,B,C) __builtin_amdgcn_wmma_f32_16x16x32_f16(false, A, false, B, (short)0, C, false, false)

__global__ __launch_bounds__(256) void moe_gemm_kernel(
    const _Float16* __restrict__ xh,    // [T, HID] f16
    const _Float16* __restrict__ wgT,   // per-expert [Ftot, HID] (B^T)
    const _Float16* __restrict__ wuT,   // per-expert [Ftot, HID]
    const _Float16* __restrict__ wdT,   // per-expert [HID, Ftot] (B^T)
    float* __restrict__ out,            // [T, HID] f32, atomic accum
    const float* __restrict__ combine,  // [T, NE]
    const int* __restrict__ lists,      // [NE, T]
    const int* __restrict__ counts,     // [NE]
    int Ftot,                           // full intermediate dim of this expert set
    int Fchunk,                         // columns of F handled by this block
    int Foff,                           // starting F column of this block's chunk
    int T, float scale, int useList)
{
    extern __shared__ char smem[];
    int*      sm_tok = (int*)smem;                  // 32 ints
    float*    sm_w   = (float*)(smem + 128);        // 32 floats
    _Float16* lds_x  = (_Float16*)(smem + 256);     // [32][HID]
    _Float16* lds_h  = lds_x + MTILE * HID;         // [32][Fchunk]

    int tilesPerE = T / MTILE;
    int e    = useList ? (int)(blockIdx.x / tilesPerE) : 0;
    int tile = useList ? (int)(blockIdx.x % tilesPerE) : (int)blockIdx.x;

    int nTok = useList ? counts[e] : T;
    if (tile * MTILE >= nTok) return;

    int tid  = threadIdx.x;
    int lane = tid & 31;
    int wave = tid >> 5;
    int hs   = lane >> 4;     // K half-select
    int ln   = lane & 15;

    if (tid < MTILE) {
        int r = tile * MTILE + tid;
        int t; float w;
        if (useList) {
            if (r < nTok) { t = lists[e * T + r]; w = combine[(size_t)t * NE + e] * scale; }
            else          { t = 0; w = 0.f; }
        } else { t = r; w = scale; }
        sm_tok[tid] = t;
        sm_w[tid]   = w;
    }
    __syncthreads();

    // stage 32 gathered x rows into LDS (uint4 = 8 halves per chunk)
    {
        const int CH = HID / 8;
        for (int c = tid; c < MTILE * CH; c += 256) {
            int m = c / CH, j = c - m * CH;
            ((uint4*)(lds_x + m * HID))[j] =
                ((const uint4*)(xh + (size_t)sm_tok[m] * HID))[j];
        }
    }
    __syncthreads();

    const _Float16* wg_e = wgT + (size_t)e * Ftot * HID;
    const _Float16* wu_e = wuT + (size_t)e * Ftot * HID;
    const _Float16* wd_e = wdT + (size_t)e * HID * Ftot;

    // phase 1: h = silu(x@Wg) * (x@Wu); each B-tile feeds both A row-tiles
    for (int nt = wave; nt < (Fchunk >> 4); nt += NWAVES) {
        v8f ag0 = {}, au0 = {}, ag1 = {}, au1 = {};
        int ncol = nt * 16 + ln;       // column within chunk
        int n    = Foff + ncol;        // global F column
        const _Float16* bg  = wg_e + (size_t)n * HID + hs * 16;
        const _Float16* bu  = wu_e + (size_t)n * HID + hs * 16;
        const _Float16* a0p = lds_x + ln * HID + hs * 16;
        const _Float16* a1p = lds_x + (16 + ln) * HID + hs * 16;
        for (int k0 = 0; k0 < HID; k0 += 32) {
            v16h a0 = load_a16(a0p + k0);
            v16h a1 = load_a16(a1p + k0);
            v16h b0 = *(const v16h*)(bg + k0);
            v16h b1 = *(const v16h*)(bu + k0);
            ag0 = WMMA16(a0, b0, ag0);
            au0 = WMMA16(a0, b1, au0);
            ag1 = WMMA16(a1, b0, ag1);
            au1 = WMMA16(a1, b1, au1);
        }
#pragma unroll
        for (int r = 0; r < 8; r++) {
            int m = r + 8 * hs;        // C/D layout: row = vgpr + 8*(lane>>4)
            float g0 = ag0[r], u0 = au0[r];
            float g1 = ag1[r], u1 = au1[r];
            lds_h[m * Fchunk + ncol]        = (_Float16)((g0 / (1.f + __expf(-g0))) * u0);
            lds_h[(16 + m) * Fchunk + ncol] = (_Float16)((g1 / (1.f + __expf(-g1))) * u1);
        }
    }
    __syncthreads();

    // phase 2: y(+=) = h @ Wd[Foff:Foff+Fchunk, :] -> scaled atomic scatter-add
    for (int nt = wave; nt < (HID >> 4); nt += NWAVES) {
        v8f acc0 = {}, acc1 = {};
        int n = nt * 16 + ln;
        const _Float16* bd  = wd_e + (size_t)n * Ftot + Foff + hs * 16;
        const _Float16* a0p = lds_h + ln * Fchunk + hs * 16;
        const _Float16* a1p = lds_h + (16 + ln) * Fchunk + hs * 16;
        for (int k0 = 0; k0 < Fchunk; k0 += 32) {
            v16h a0 = load_a16(a0p + k0);
            v16h a1 = load_a16(a1p + k0);
            v16h b  = *(const v16h*)(bd + k0);
            acc0 = WMMA16(a0, b, acc0);
            acc1 = WMMA16(a1, b, acc1);
        }
#pragma unroll
        for (int r = 0; r < 8; r++) {
            int m  = r + 8 * hs;
            float w0 = sm_w[m];
            float w1 = sm_w[16 + m];
            if (w0 != 0.f) unsafeAtomicAdd(&out[(size_t)sm_tok[m]      * HID + n], acc0[r] * w0);
            if (w1 != 0.f) unsafeAtomicAdd(&out[(size_t)sm_tok[16 + m] * HID + n], acc1[r] * w1);
        }
    }
}

// ---------------- host ----------------

extern "C" void kernel_launch(void* const* d_in, const int* in_sizes, int n_in,
                              void* d_out, int out_size, void* d_ws, size_t ws_size,
                              hipStream_t stream) {
    const float* x   = (const float*)d_in[0];
    const float* wr  = (const float*)d_in[1];
    const float* wg  = (const float*)d_in[2];
    const float* wu  = (const float*)d_in[3];
    const float* wd  = (const float*)d_in[4];
    const float* wsg = (const float*)d_in[5];
    const float* wsu = (const float*)d_in[6];
    const float* wsd = (const float*)d_in[7];
    float* out = (float*)d_out;

    const int T  = in_sizes[0] / HID;          // 2048
    const int F  = in_sizes[2] / (NE * HID);   // 1408
    const int FS = in_sizes[5] / HID;          // 2816

    // workspace partition (256B aligned)
    char*  ws  = (char*)d_ws;
    size_t off = 0;
    auto alloc = [&](size_t bytes) -> char* {
        char* p = ws + off;
        off += (bytes + 255) & ~(size_t)255;
        return p;
    };
    _Float16* xh      = (_Float16*)alloc((size_t)T * HID * 2);
    float*    combine = (float*)   alloc((size_t)T * NE * 4);
    int*      counts  = (int*)     alloc(256);
    int*      lists   = (int*)     alloc((size_t)NE * T * 4);
    _Float16* wgT     = (_Float16*)alloc((size_t)NE * F * HID * 2);
    _Float16* wuT     = (_Float16*)alloc((size_t)NE * F * HID * 2);
    _Float16* wdT     = (_Float16*)alloc((size_t)NE * F * HID * 2);
    _Float16* wsgT    = (_Float16*)alloc((size_t)FS * HID * 2);
    _Float16* wsuT    = (_Float16*)alloc((size_t)FS * HID * 2);
    _Float16* wsdT    = (_Float16*)alloc((size_t)FS * HID * 2);

    hipMemsetAsync(out, 0, (size_t)T * HID * 4, stream);
    hipMemsetAsync(counts, 0, 256, stream);

    // activations -> f16
    {
        int n = T * HID;
        cvt_x_f16<<<(n + 255) / 256, 256, 0, stream>>>(x, xh, n);
    }
    // weights -> f16, transposed to [N, K]
    {
        int n1 = HID * F;
        transpose_cvt<<<dim3((n1 + 255) / 256, NE), 256, 0, stream>>>(wg, wgT, HID, F);
        transpose_cvt<<<dim3((n1 + 255) / 256, NE), 256, 0, stream>>>(wu, wuT, HID, F);
        transpose_cvt<<<dim3((n1 + 255) / 256, NE), 256, 0, stream>>>(wd, wdT, F, HID);
        int n2 = HID * FS;
        transpose_cvt<<<dim3((n2 + 255) / 256, 1), 256, 0, stream>>>(wsg, wsgT, HID, FS);
        transpose_cvt<<<dim3((n2 + 255) / 256, 1), 256, 0, stream>>>(wsu, wsuT, HID, FS);
        transpose_cvt<<<dim3((n2 + 255) / 256, 1), 256, 0, stream>>>(wsd, wsdT, FS, HID);
    }

    router_kernel<<<T, 64, 0, stream>>>(x, wr, combine);
    build_lists<<<(T + 255) / 256, 256, 0, stream>>>(combine, counts, lists, T);

    // LDS: header + 32xHID x-tile + 32xFchunk h-tile  (Fchunk = 1408 in all cases)
    const int FCH = F;                   // 1408 == FS/2
    size_t lds_bytes = 256 + (size_t)MTILE * HID * 2 + (size_t)MTILE * FCH * 2; // ~216 KB
    hipFuncSetAttribute(reinterpret_cast<const void*>(moe_gemm_kernel),
                        hipFuncAttributeMaxDynamicSharedMemorySize, (int)lds_bytes);

    // routed experts: grid = E * (T/32) tiles, early-exit past each expert's count
    moe_gemm_kernel<<<NE * (T / MTILE), 256, lds_bytes, stream>>>(
        xh, wgT, wuT, wdT, out, combine, lists, counts,
        F, F, 0, T, SCALE_F, 1);

    // shared expert: identity token map, weight 1.0; F split into two 1408-col halves
    moe_gemm_kernel<<<T / MTILE, 256, lds_bytes, stream>>>(
        xh, wsgT, wsuT, wsdT, out, combine, nullptr, nullptr,
        FS, FS / 2, 0, T, 1.0f, 0);
    moe_gemm_kernel<<<T / MTILE, 256, lds_bytes, stream>>>(
        xh, wsgT, wsuT, wsdT, out, combine, nullptr, nullptr,
        FS, FS / 2, FS / 2, T, 1.0f, 0);
}
